// GIN_15616501088594
// MI455X (gfx1250) — matrix operand order repaired and checked by reading, not compiled, not run on hardware
//
#include <hip/hip_runtime.h>

typedef __attribute__((ext_vector_type(16))) _Float16 v16h;
typedef __attribute__((ext_vector_type(8)))  float    v8f;

#define GIN_N     100000
#define GIN_E     1600000
#define GIN_G     256
#define GIN_DIN   64
#define GIN_H     128
#define GIN_BNEPS 1e-5f
#define BM        32

// ---------------- small helpers ----------------
__global__ void k_cvt_f16(const float* __restrict__ in, _Float16* __restrict__ out, int n) {
  for (int i = blockIdx.x * blockDim.x + threadIdx.x; i < n; i += gridDim.x * blockDim.x)
    out[i] = (_Float16)in[i];
}

__global__ void k_copy_f32(const float* __restrict__ in, float* __restrict__ out, int n) {
  for (int i = blockIdx.x * blockDim.x + threadIdx.x; i < n; i += gridDim.x * blockDim.x)
    out[i] = in[i];
}

// ---------------- edge scatter-add (L2-resident gather + f32 atomics) ----------------
__global__ void k_scatter_add(const float* __restrict__ hsrc,
                              const int* __restrict__ src,
                              const int* __restrict__ dst,
                              float* agg, unsigned total, int kshift) {
  const unsigned kmask = (1u << kshift) - 1u;
  const unsigned stride = gridDim.x * blockDim.x;
  for (unsigned i = blockIdx.x * blockDim.x + threadIdx.x; i < total; i += stride) {
    unsigned e = i >> kshift;
    unsigned c = i & kmask;
    unsigned s = (unsigned)src[e];
    unsigned d = (unsigned)dst[e];
    __builtin_prefetch(&src[e + (stride >> kshift)], 0, 0);  // global_prefetch
    atomicAdd(&agg[(d << kshift) + c], hsrc[(s << kshift) + c]);
  }
}

// ---------------- fused GEMM + bias + ReLU (+ BN stats) via WMMA ----------------
// Block: 256 threads (8 waves), computes 32 rows x 128 cols.
// Wave w -> cols [16w,16w+16); two 16x16 M-subtiles per wave.
template<bool AF16, bool OUTF16, bool STATS>
__global__ __launch_bounds__(256) void k_gin_gemm(
    const void* __restrict__ Ain, const _Float16* __restrict__ W,
    const float* __restrict__ bias, void* __restrict__ Out,
    float* colsum, float* colsumsq, int K) {
  __shared__ __align__(16) _Float16 sA[BM][GIN_H + 8];       // [row][k], padded
  __shared__ __align__(16) _Float16 sB[GIN_H][GIN_H + 8];    // [col][k] (W transposed), padded
  const int tid = threadIdx.x;
  const int row0 = blockIdx.x * BM;
  const int kshift = (K == 64) ? 6 : 7;

  // Stage A tile (convert f32->f16 when needed) and W transposed into LDS
  const int elems = BM * K;
  if (AF16) {
    const _Float16* A = (const _Float16*)Ain;
    for (int i = tid; i < elems; i += 256) {
      int r = i >> kshift, k = i & (K - 1);
      sA[r][k] = A[(size_t)(row0 + r) * K + k];
    }
  } else {
    const float* A = (const float*)Ain;
    for (int i = tid; i < elems; i += 256) {
      int r = i >> kshift, k = i & (K - 1);
      sA[r][k] = (_Float16)A[(size_t)(row0 + r) * K + k];
    }
  }
  for (int i = tid; i < K * GIN_H; i += 256) {
    int k = i >> 7, c = i & 127;   // W row-major [K][128]
    sB[c][k] = W[i];
  }
  __syncthreads();

  const int lane = tid & 31;
  const int col0 = (tid >> 5) * 16;
  const int hi   = (lane >> 4) & 1;   // 0: lanes 0-15, 1: lanes 16-31
  const int rA   = lane & 15;

  v8f acc[2] = {};
  for (int k0 = 0; k0 < K; k0 += 32) {
    union { v16h v; uint4 q[2]; } a0, a1, b;
    // A frag: halves 0..7 = K[k0+8*hi .. +7], halves 8..15 = K[k0+16+8*hi .. +7]
    const int ka = k0 + hi * 8;
    a0.q[0] = *(const uint4*)&sA[rA][ka];
    a0.q[1] = *(const uint4*)&sA[rA][ka + 16];
    a1.q[0] = *(const uint4*)&sA[16 + rA][ka];
    a1.q[1] = *(const uint4*)&sA[16 + rA][ka + 16];
    // B frag: halves 0..7 = K[k0+8*hi .. +7], halves 8..15 = K[k0+16+8*hi .. +7], column = col0+rA
    const int kb = k0 + hi * 8;
    b.q[0] = *(const uint4*)&sB[col0 + rA][kb];
    b.q[1] = *(const uint4*)&sB[col0 + rA][kb + 16];
    acc[0] = __builtin_amdgcn_wmma_f32_16x16x32_f16(false, a0.v, false, b.v,
                                                    (short)0, acc[0], false, false);
    acc[1] = __builtin_amdgcn_wmma_f32_16x16x32_f16(false, a1.v, false, b.v,
                                                    (short)0, acc[1], false, false);
  }

  // Epilogue: bias + ReLU (+ column stats for BN)
  const int cc = col0 + rA;
  const float bv = bias[cc];
  float s1 = 0.f, s2 = 0.f;
#pragma unroll
  for (int m = 0; m < 2; ++m) {
#pragma unroll
    for (int v = 0; v < 8; ++v) {
      float val = fmaxf(acc[m][v] + bv, 0.f);
      int r = row0 + m * 16 + hi * 8 + v;   // D layout: M = v + 8*hi
      if (OUTF16) ((_Float16*)Out)[(size_t)r * GIN_H + cc] = (_Float16)val;
      else        ((float*)Out)[(size_t)r * GIN_H + cc] = val;
      if (STATS) { s1 += val; s2 += val * val; }
    }
  }
  if (STATS) {
    atomicAdd(&colsum[cc], s1);
    atomicAdd(&colsumsq[cc], s2);
  }
}

// ---------------- BatchNorm ----------------
__global__ void k_bn_finalize(const float* __restrict__ colsum,
                              const float* __restrict__ colsumsq,
                              const float* __restrict__ g, const float* __restrict__ be,
                              float* __restrict__ scale, float* __restrict__ shift) {
  int c = threadIdx.x;                  // 128 threads
  const float invN = 1.0f / (float)GIN_N;
  float mu  = colsum[c] * invN;
  float var = colsumsq[c] * invN - mu * mu;
  float sc  = g[c] * rsqrtf(var + GIN_BNEPS);
  scale[c] = sc;
  shift[c] = be[c] - mu * sc;
}

// z -> relu(z*scale+shift); writes h and next-layer agg init (may alias z in-place)
__global__ void k_bn_apply(const float* z, const float* __restrict__ scale,
                           const float* __restrict__ shift,
                           float* hout, float* aggout, unsigned total) {
  for (unsigned i = blockIdx.x * blockDim.x + threadIdx.x; i < total;
       i += gridDim.x * blockDim.x) {
    int c = i & (GIN_H - 1);
    float v = fmaxf(fmaf(z[i], scale[c], shift[c]), 0.f);
    hout[i] = v;
    aggout[i] = v;
  }
}

// ---------------- pooling + output projection ----------------
__global__ void k_pool(const float* __restrict__ h, const int* __restrict__ batch,
                       float* pooled, unsigned total) {
  for (unsigned i = blockIdx.x * blockDim.x + threadIdx.x; i < total;
       i += gridDim.x * blockDim.x) {
    unsigned n = i >> 7, c = i & 127;
    atomicAdd(&pooled[((unsigned)batch[n] << 7) + c], h[i]);
  }
}

__global__ void k_out(const float* __restrict__ pooled, const float* __restrict__ Wout,
                      const float* __restrict__ bout, float* __restrict__ out) {
  int g = threadIdx.x;                  // 256 threads, one graph each
  float s = bout[0];
  for (int c = 0; c < GIN_H; ++c) s = fmaf(pooled[g * GIN_H + c], Wout[c], s);
  out[g] = s;
}

// ---------------- host orchestration ----------------
extern "C" void kernel_launch(void* const* d_in, const int* in_sizes, int n_in,
                              void* d_out, int out_size, void* d_ws, size_t ws_size,
                              hipStream_t stream) {
  (void)in_sizes; (void)n_in; (void)out_size; (void)ws_size;
  const float* x     = (const float*)d_in[0];
  const int*   ei    = (const int*)d_in[1];
  const int*   batch = (const int*)d_in[2];
  const float* W1_0  = (const float*)d_in[3];
  const float* b1_0  = (const float*)d_in[4];
  const float* W2_0  = (const float*)d_in[5];
  const float* b2_0  = (const float*)d_in[6];
  const float* g_0   = (const float*)d_in[7];
  const float* be_0  = (const float*)d_in[8];
  const float* W1s   = (const float*)d_in[9];
  const float* b1s   = (const float*)d_in[10];
  const float* W2s   = (const float*)d_in[11];
  const float* b2s   = (const float*)d_in[12];
  const float* gs    = (const float*)d_in[13];
  const float* bes   = (const float*)d_in[14];
  const float* Wout  = (const float*)d_in[15];
  const float* bout  = (const float*)d_in[16];
  const int* src = ei;
  const int* dst = ei + GIN_E;

  // workspace carve-out
  char* ws = (char*)d_ws;
  size_t off = 0;
  auto carve = [&](size_t bytes) -> void* {
    void* p = ws + off;
    off = (off + bytes + 255) & ~(size_t)255;
    return p;
  };
  float*    bufA  = (float*)carve((size_t)GIN_N * GIN_H * 4);     // agg / z2
  float*    bufH  = (float*)carve((size_t)GIN_N * GIN_H * 4);     // h
  _Float16* bufZ1 = (_Float16*)carve((size_t)GIN_N * GIN_H * 2);  // relu(z@W1+b1), f16
  _Float16* w16   = (_Float16*)carve((size_t)90112 * 2);          // f16 weights
  float* colsum   = (float*)carve(512);
  float* colsumsq = (float*)carve(512);
  float* scale    = (float*)carve(512);
  float* shift    = (float*)carve(512);
  float* pooled   = (float*)carve((size_t)GIN_G * GIN_H * 4);

  _Float16* w_W1_0 = w16;
  _Float16* w_W2_0 = w16 + 8192;
  _Float16* w_W1s  = w16 + 8192 + 16384;           // 2 x 128x128
  _Float16* w_W2s  = w_W1s + 32768;

  k_cvt_f16<<<32, 256, 0, stream>>>(W1_0, w_W1_0, 8192);
  k_cvt_f16<<<64, 256, 0, stream>>>(W2_0, w_W2_0, 16384);
  k_cvt_f16<<<128, 256, 0, stream>>>(W1s, w_W1s, 32768);
  k_cvt_f16<<<128, 256, 0, stream>>>(W2s, w_W2s, 32768);

  const int GBIG = 32768;
  for (int l = 0; l < 3; ++l) {
    const int K = (l == 0) ? GIN_DIN : GIN_H;
    const int kshift = (l == 0) ? 6 : 7;
    const float* hsrc = (l == 0) ? x : bufH;
    if (l == 0)
      k_copy_f32<<<8192, 256, 0, stream>>>(x, bufA, GIN_N * GIN_DIN);  // agg := x

    k_scatter_add<<<GBIG, 256, 0, stream>>>(hsrc, src, dst, bufA,
                                            (unsigned)GIN_E << kshift, kshift);

    const _Float16* W1 = (l == 0) ? w_W1_0 : (w_W1s + (size_t)(l - 1) * 16384);
    const float*    b1 = (l == 0) ? b1_0   : (b1s + (l - 1) * GIN_H);
    const _Float16* W2 = (l == 0) ? w_W2_0 : (w_W2s + (size_t)(l - 1) * 16384);
    const float*    b2 = (l == 0) ? b2_0   : (b2s + (l - 1) * GIN_H);
    const float*    gg = (l == 0) ? g_0    : (gs + (l - 1) * GIN_H);
    const float*    bb = (l == 0) ? be_0   : (bes + (l - 1) * GIN_H);

    // z1 = relu(agg @ W1 + b1)  [f32 in, f16 out]
    k_gin_gemm<false, true, false><<<GIN_N / BM, 256, 0, stream>>>(
        bufA, W1, b1, bufZ1, nullptr, nullptr, K);
    // z2 = relu(z1 @ W2 + b2)   [f16 in, f32 out + BN stats]
    hipMemsetAsync(colsum, 0, 1024, stream);   // colsum + colsumsq contiguous
    k_gin_gemm<true, false, true><<<GIN_N / BM, 256, 0, stream>>>(
        bufZ1, W2, b2, bufA, colsum, colsumsq, GIN_H);
    // BN + outer ReLU; writes h and re-initializes agg (in place on z2) for next layer
    k_bn_finalize<<<1, GIN_H, 0, stream>>>(colsum, colsumsq, gg, bb, scale, shift);
    k_bn_apply<<<GBIG, 256, 0, stream>>>(bufA, scale, shift, bufH, bufA,
                                         (unsigned)GIN_N * GIN_H);
  }

  hipMemsetAsync(pooled, 0, (size_t)GIN_G * GIN_H * 4, stream);
  k_pool<<<GBIG, 256, 0, stream>>>(bufH, batch, pooled, (unsigned)GIN_N * GIN_H);
  k_out<<<1, GIN_G, 0, stream>>>(pooled, Wout, bout, (float*)d_out);
}
